// GraphAttentionLayer_78769700208701
// MI455X (gfx1250) — matrix operand order repaired and checked by reading, not compiled, not run on hardware
//
#include <hip/hip_runtime.h>

// GAT layer: B=1, N=4096, F=128, D=64, H=4.
// Phase 1: h = feat @ W (fp32, exact), store hT as f16 in B-fragment-friendly
//          layout; reduce a_self/a_neigh; per-head max(a_neigh) via uint atomicMax.
// Phase 2: flash-style masked softmax + (P @ h) via v_wmma_f32_16x16x32_f16.

#define GN 4096
#define GF 128
#define GD 64
#define GH 4

typedef __attribute__((ext_vector_type(16))) _Float16 v16h;
typedef __attribute__((ext_vector_type(8)))  _Float16 v8h;
typedef __attribute__((ext_vector_type(8)))  float    v8f;

__global__ __launch_bounds__(32) void gat_init(unsigned* __restrict__ nmax) {
  if (threadIdx.x < GH) nmax[threadIdx.x] = 0u;  // monotone-uint encoding of -inf
}

// ---------------- Phase 1: projection + attention coefficients ----------------
__global__ __launch_bounds__(256) void gat_phase1(
    const float* __restrict__ feat,    // [N,F]
    const float* __restrict__ W,       // [H,F,D]
    const float* __restrict__ ak,      // [H,2D]
    _Float16*    __restrict__ hT,      // [H,D,N]  (f16, transposed)
    float*       __restrict__ a_self,  // [H,N]
    float*       __restrict__ a_neigh, // [H,N]
    unsigned*    __restrict__ nmax)    // [H] monotone-uint max of a_neigh
{
  __shared__ float sf[GF];
  __shared__ float red[512];
  const int i = blockIdx.x;
  const int t = threadIdx.x;
  if (t < GF) sf[t] = feat[i * GF + t];
  __syncthreads();

  const int hd = t >> 6, d = t & 63;
  const float* wp = W + (size_t)hd * GF * GD + d;
  float acc = 0.f;
  #pragma unroll 8
  for (int k = 0; k < GF; ++k) acc = __builtin_fmaf(sf[k], wp[(size_t)k * GD], acc);

  hT[(size_t)(hd * GD + d) * GN + i] = (_Float16)acc;

  red[t]       = acc * ak[hd * 2 * GD + d];        // a_self partial
  red[256 + t] = acc * ak[hd * 2 * GD + GD + d];   // a_neigh partial
  __syncthreads();
  #pragma unroll
  for (int off = 32; off > 0; off >>= 1) {
    if (d < off) { red[t] += red[t + off]; red[256 + t] += red[256 + t + off]; }
    __syncthreads();
  }
  if (d == 0) {
    float as = red[t], an = red[256 + t];
    a_self[hd * GN + i]  = as;
    a_neigh[hd * GN + i] = an;
    unsigned b = __float_as_uint(an);
    unsigned u = (b & 0x80000000u) ? ~b : (b | 0x80000000u);
    atomicMax(&nmax[hd], u);
  }
}

// ---------------- Phase 2: masked softmax + WMMA apply ----------------
__device__ __forceinline__ float gat_p(int a, float an, float as, float m) {
  float s = as + an;
  s = (s > 0.f) ? s : 0.2f * s;                    // leaky_relu(0.2)
  return a ? __expf(s - m) : 0.f;                  // adjacency mask
}

__global__ __launch_bounds__(128) void gat_phase2(
    const int*      __restrict__ adj,      // [N,N]
    const _Float16* __restrict__ hT,       // [H,D,N]
    const float*    __restrict__ a_self,   // [H,N]
    const float*    __restrict__ a_neigh,  // [H,N]
    const unsigned* __restrict__ nmax,     // [H]
    float*          __restrict__ out)      // [N, H*D]
{
  const int lane = threadIdx.x & 31;
  const int gw   = blockIdx.x * 4 + (threadIdx.x >> 5);
  const int hd   = gw & (GH - 1);          // head
  const int rb   = gw >> 2;                // 16-row block, 0..255
  const int mr   = lane & 15;              // A-frag row / B-frag col / D-frag col
  const int kg   = lane >> 4;              // K half-group
  const int i    = rb * 16 + mr;

  // stability bound: m >= leaky(a_self_i + a_neigh_j) for all j (leaky monotone)
  unsigned u = nmax[hd];
  float nm = __uint_as_float((u & 0x80000000u) ? (u ^ 0x80000000u) : ~u);
  float as = a_self[hd * GN + i];
  float sm = as + nm;
  float m  = (sm > 0.f) ? sm : 0.2f * sm;

  const int*      ar = adj + (size_t)i * GN;
  const float*    an = a_neigh + (size_t)hd * GN;
  const _Float16* hb = hT + (size_t)hd * GD * GN;

  v8f c0 = {}, c1 = {}, c2 = {}, c3 = {};
  float rs = 0.f;                           // softmax denominator (partial)

  for (int k0 = 0; k0 < GN; k0 += 32) {
    // A-fragment lane layout (16-bit A 16x32): lanes 0-15 -> K {0..7,16..23},
    // lanes 16-31 -> K {8..15,24..31}, row M = lane&15.
    const int lo = k0 + kg * 8;
    const int hi = lo + 16;
    if (k0 + 128 < GN) __builtin_prefetch((const void*)(ar + k0 + 128), 0, 1);

    int4   A0 = *(const int4*)(ar + lo);
    int4   A1 = *(const int4*)(ar + lo + 4);
    int4   A2 = *(const int4*)(ar + hi);
    int4   A3 = *(const int4*)(ar + hi + 4);
    float4 N0 = *(const float4*)(an + lo);
    float4 N1 = *(const float4*)(an + lo + 4);
    float4 N2 = *(const float4*)(an + hi);
    float4 N3 = *(const float4*)(an + hi + 4);

    float p[16];
    p[ 0]=gat_p(A0.x,N0.x,as,m); p[ 1]=gat_p(A0.y,N0.y,as,m);
    p[ 2]=gat_p(A0.z,N0.z,as,m); p[ 3]=gat_p(A0.w,N0.w,as,m);
    p[ 4]=gat_p(A1.x,N1.x,as,m); p[ 5]=gat_p(A1.y,N1.y,as,m);
    p[ 6]=gat_p(A1.z,N1.z,as,m); p[ 7]=gat_p(A1.w,N1.w,as,m);
    p[ 8]=gat_p(A2.x,N2.x,as,m); p[ 9]=gat_p(A2.y,N2.y,as,m);
    p[10]=gat_p(A2.z,N2.z,as,m); p[11]=gat_p(A2.w,N2.w,as,m);
    p[12]=gat_p(A3.x,N3.x,as,m); p[13]=gat_p(A3.y,N3.y,as,m);
    p[14]=gat_p(A3.z,N3.z,as,m); p[15]=gat_p(A3.w,N3.w,as,m);

    v16h Af;
    #pragma unroll
    for (int e = 0; e < 16; ++e) { rs += p[e]; Af[e] = (_Float16)p[e]; }

    // B-fragment (16-bit B 32x16): lane holds K {8kg..8kg+7, 16+8kg..16+8kg+7} for col N=lane&15
    // -> two b128 loads from the transposed hT layout.
    const _Float16* bp = hb + (size_t)mr * GN + k0 + kg * 8;
    v8h b0l = *(const v8h*)(bp);
    v8h b0h = *(const v8h*)(bp + 16);
    v8h b1l = *(const v8h*)(bp + 16 * GN);
    v8h b1h = *(const v8h*)(bp + 16 * GN + 16);
    v8h b2l = *(const v8h*)(bp + 32 * GN);
    v8h b2h = *(const v8h*)(bp + 32 * GN + 16);
    v8h b3l = *(const v8h*)(bp + 48 * GN);
    v8h b3h = *(const v8h*)(bp + 48 * GN + 16);
    v16h B0 = __builtin_shufflevector(b0l, b0h, 0,1,2,3,4,5,6,7,8,9,10,11,12,13,14,15);
    v16h B1 = __builtin_shufflevector(b1l, b1h, 0,1,2,3,4,5,6,7,8,9,10,11,12,13,14,15);
    v16h B2 = __builtin_shufflevector(b2l, b2h, 0,1,2,3,4,5,6,7,8,9,10,11,12,13,14,15);
    v16h B3 = __builtin_shufflevector(b3l, b3h, 0,1,2,3,4,5,6,7,8,9,10,11,12,13,14,15);

    c0 = __builtin_amdgcn_wmma_f32_16x16x32_f16(false, Af, false, B0, (short)0, c0, false, false);
    c1 = __builtin_amdgcn_wmma_f32_16x16x32_f16(false, Af, false, B1, (short)0, c1, false, false);
    c2 = __builtin_amdgcn_wmma_f32_16x16x32_f16(false, Af, false, B2, (short)0, c2, false, false);
    c3 = __builtin_amdgcn_wmma_f32_16x16x32_f16(false, Af, false, B3, (short)0, c3, false, false);
  }

  // lanes l and l+16 hold complementary K halves of row (l&15): combine.
  rs += __shfl_xor(rs, 16, 32);
  float inv = (rs > 0.f) ? 1.f / rs : 0.f;   // fully-masked row -> 0 (attn*adj)

  // D-frag: VGPR r -> row (r + 8*kg), col N = lane&15. Broadcast the right 1/sum.
  float* ob = out + (size_t)(rb * 16) * (GH * GD) + hd * GD + mr;
  #pragma unroll
  for (int r = 0; r < 8; ++r) {
    float ir = __shfl(inv, r + 8 * kg, 32);
    float* op = ob + (size_t)(r + 8 * kg) * (GH * GD);
    float v0 = c0[r] * ir; op[ 0] = v0 > 0.f ? v0 : 0.f;   // relu
    float v1 = c1[r] * ir; op[16] = v1 > 0.f ? v1 : 0.f;
    float v2 = c2[r] * ir; op[32] = v2 > 0.f ? v2 : 0.f;
    float v3 = c3[r] * ir; op[48] = v3 > 0.f ? v3 : 0.f;
  }
}

extern "C" void kernel_launch(void* const* d_in, const int* in_sizes, int n_in,
                              void* d_out, int out_size, void* d_ws, size_t ws_size,
                              hipStream_t stream) {
  const int*   adj  = (const int*)d_in[0];
  const float* feat = (const float*)d_in[1];
  const float* W    = (const float*)d_in[2];
  const float* ak   = (const float*)d_in[3];
  float*       out  = (float*)d_out;

  // workspace layout (~2.2 MB total)
  char* ws = (char*)d_ws;
  _Float16* hT      = (_Float16*)ws;                                  // 2 MB
  float*    a_self  = (float*)(ws + (size_t)(2u << 20));              // 64 KB
  float*    a_neigh = (float*)(ws + (size_t)(2u << 20) + 65536);      // 64 KB
  unsigned* nmax    = (unsigned*)(ws + (size_t)(2u << 20) + 2 * 65536);

  gat_init  <<<1, 32, 0, stream>>>(nmax);
  gat_phase1<<<GN, 256, 0, stream>>>(feat, W, ak, hT, a_self, a_neigh, nmax);
  gat_phase2<<<(GN / 16) * GH / 4, 128, 0, stream>>>(adj, hT, a_self, a_neigh, nmax, out);
}
